// TransformerEncoderLayerSimple_66048007078350
// MI455X (gfx1250) — compile-verified, matrix operand order checked by reading, not run on hardware
//
#include <hip/hip_runtime.h>
#include <math.h>

// Problem constants (from reference): S=4096, N=32, E=512, H=8, DK=64, l=2048.
#define SS 4096
#define NN 32
#define EE 512
#define HH 8

typedef __attribute__((ext_vector_type(16))) __bf16 v16bf;
typedef __attribute__((ext_vector_type(8)))  float  v8f;

// ---------------------------------------------------------------------------
// Fragment loaders for v_wmma_f32_16x16x32_bf16.
// 16-bit 16x32 A-matrix layout (ISA 7.12.2): lane holds row m=lane&15;
// half=lane>>4 selects K runs [8*half, 8*half+8) and [16+8*half, 16+8*half+8).
// Element i of the v16: i<8 -> run0[i], i>=8 -> run1[i-8]. B mirrors A with
// M<->N (lane holds column). So a lane's fragment = two contiguous 8-float
// runs when K is the contiguous memory dimension.
// ---------------------------------------------------------------------------
__device__ __forceinline__ v16bf frag_contig(const float* __restrict__ base, int half) {
    const float* r0 = base + half * 8;
    const float* r1 = base + 16 + half * 8;
    float4 x0 = *(const float4*)(r0);
    float4 x1 = *(const float4*)(r0 + 4);
    float4 y0 = *(const float4*)(r1);
    float4 y1 = *(const float4*)(r1 + 4);
    v16bf v;
    v[0] = (__bf16)x0.x;  v[1] = (__bf16)x0.y;  v[2] = (__bf16)x0.z;  v[3] = (__bf16)x0.w;
    v[4] = (__bf16)x1.x;  v[5] = (__bf16)x1.y;  v[6] = (__bf16)x1.z;  v[7] = (__bf16)x1.w;
    v[8] = (__bf16)y0.x;  v[9] = (__bf16)y0.y;  v[10] = (__bf16)y0.z; v[11] = (__bf16)y0.w;
    v[12] = (__bf16)y1.x; v[13] = (__bf16)y1.y; v[14] = (__bf16)y1.z; v[15] = (__bf16)y1.w;
    return v;
}

// B fragment where K = s (stride N*E floats between consecutive K) and the
// lane's column is a fixed e. src layout: src[s][n][e], contiguous in e.
__device__ __forceinline__ v16bf frag_src_strideS(const float* __restrict__ src,
                                                  int sbase, int n, int e, int half) {
    v16bf v;
#pragma unroll
    for (int i = 0; i < 16; ++i) {
        int koff = (i < 8) ? (half * 8 + i) : (16 + half * 8 + (i - 8));
        size_t s = (size_t)(sbase + koff);
        v[i] = (__bf16)src[(s * NN + n) * EE + e];
    }
    return v;
}

__device__ __forceinline__ v8f wmma_bf16(v16bf a, v16bf b, v8f c) {
    return __builtin_amdgcn_wmma_f32_16x16x32_bf16(false, a, false, b, (short)0, c, false, false);
}

// ---------------------------------------------------------------------------
// Stage 1a: q[n,j] = src[l=2048,n,:] . Wq[j,:] + bq[j]      (32 x 512)
// ---------------------------------------------------------------------------
__global__ __launch_bounds__(256) void qproj_kernel(const float* __restrict__ src,
                                                    const float* __restrict__ Wq,
                                                    const float* __restrict__ bq,
                                                    float* __restrict__ q) {
    int idx = blockIdx.x * 256 + threadIdx.x;     // 16384 outputs
    int n = idx >> 9, j = idx & 511;
    const float* srow = src + ((size_t)2048 * NN + n) * EE;
    const float* wrow = Wq + (size_t)j * EE;
    float acc = bq[j];
#pragma unroll 4
    for (int e = 0; e < EE; ++e) acc = fmaf(srow[e], wrow[e], acc);
    q[idx] = acc;
}

// ---------------------------------------------------------------------------
// Stage 1b: qk[n,h,e] = sum_d q[n,h*64+d] * Wk[h*64+d,e]; h padded to 16 rows
// (pad rows written as zero so WMMA A reads are valid & exact).
// Also sb[n,h] = q[n,h,:] . bk[h,:]  (score bias from k's bias).
// ---------------------------------------------------------------------------
__global__ __launch_bounds__(256) void qk_kernel(const float* __restrict__ q,
                                                 const float* __restrict__ Wk,
                                                 const float* __restrict__ bk,
                                                 float* __restrict__ qk,
                                                 float* __restrict__ sb) {
    int idx = blockIdx.x * 256 + threadIdx.x;     // 32*16*512 = 262144
    int e = idx & 511;
    int h = (idx >> 9) & 15;
    int n = idx >> 13;
    float acc = 0.0f;
    if (h < HH) {
        const float* qv = q + (size_t)n * EE + h * 64;
#pragma unroll 8
        for (int d = 0; d < 64; ++d)
            acc = fmaf(qv[d], Wk[(size_t)(h * 64 + d) * EE + e], acc);
        if (e == 0) {
            float s = 0.0f;
            for (int d = 0; d < 64; ++d) s = fmaf(qv[d], bk[h * 64 + d], s);
            sb[n * HH + h] = s;
        }
    }
    qk[idx] = acc;
}

// ---------------------------------------------------------------------------
// Stage 2: scores[n,h,s] = qk[n,h,:] . src[s,n,:]   via WMMA bf16.
// Grid (n, s-chunk of 256). 8 waves; each wave owns two 16-wide s tiles.
// A = qk rows (M=h padded 16, K=e), B = src columns (N=s, K=e contiguous).
// ---------------------------------------------------------------------------
__global__ __launch_bounds__(256) void scores_kernel(const float* __restrict__ src,
                                                     const float* __restrict__ qk,
                                                     float* __restrict__ scores) {
    int n = blockIdx.x;            // 0..31
    int sc = blockIdx.y;           // 0..15  (256 s each)
    int wave = threadIdx.x >> 5;
    int lane = threadIdx.x & 31;
    int half = lane >> 4;
    int m = lane & 15;
    int s0 = sc * 256;
    int colA = s0 + (wave * 2 + 0) * 16 + m;    // B tile0 column (this lane)
    int colB = s0 + (wave * 2 + 1) * 16 + m;    // B tile1 column
    const float* qkrow = qk + ((size_t)n * 16 + m) * EE;     // padded rows
    const float* srcA = src + ((size_t)colA * NN + n) * EE;
    const float* srcB = src + ((size_t)colB * NN + n) * EE;

    v8f acc0 = {}, acc1 = {};
#pragma unroll
    for (int ke = 0; ke < 16; ++ke) {
        int e0 = ke * 32;
        v16bf a  = frag_contig(qkrow + e0, half);
        v16bf b0 = frag_contig(srcA + e0, half);
        v16bf b1 = frag_contig(srcB + e0, half);
        acc0 = wmma_bf16(a, b0, acc0);
        acc1 = wmma_bf16(a, b1, acc1);
    }
    // C layout: lanes 0-15 -> M=r (real heads 0..7), N=lane. Lanes 16-31 are pad rows.
    if (half == 0) {
#pragma unroll
        for (int r = 0; r < 8; ++r) {
            scores[((size_t)(n * HH + r)) * SS + (size_t)(s0 + (wave * 2 + 0) * 16 + m)] = acc0[r];
            scores[((size_t)(n * HH + r)) * SS + (size_t)(s0 + (wave * 2 + 1) * 16 + m)] = acc1[r];
        }
    }
}

// ---------------------------------------------------------------------------
// Stage 3: softmax over s (4096) per (n,h); writes p padded to 16 heads
// (zero pad rows) so it can feed WMMA A directly.  x = (raw + sb)/sqrt(64).
// ---------------------------------------------------------------------------
__global__ __launch_bounds__(256) void softmax_kernel(const float* __restrict__ scores,
                                                      const float* __restrict__ sb,
                                                      float* __restrict__ p) {
    int nh16 = blockIdx.x;               // 0..511  (n*16+h)
    int n = nh16 >> 4, h = nh16 & 15;
    int t = threadIdx.x;
    float* prow = p + (size_t)nh16 * SS;
    if (h >= HH) {
#pragma unroll
        for (int k = 0; k < 16; ++k) prow[t + k * 256] = 0.0f;
        return;
    }
    const float* srow = scores + ((size_t)(n * HH + h)) * SS;
    float bias = sb[n * HH + h];
    float x[16];
    float mx = -INFINITY;
#pragma unroll
    for (int k = 0; k < 16; ++k) {
        x[k] = (srow[t + k * 256] + bias) * 0.125f;
        mx = fmaxf(mx, x[k]);
    }
    __shared__ float red[256];
    red[t] = mx; __syncthreads();
    for (int off = 128; off > 0; off >>= 1) {
        if (t < off) red[t] = fmaxf(red[t], red[t + off]);
        __syncthreads();
    }
    mx = red[0]; __syncthreads();
    float sum = 0.0f;
#pragma unroll
    for (int k = 0; k < 16; ++k) { x[k] = __expf(x[k] - mx); sum += x[k]; }
    red[t] = sum; __syncthreads();
    for (int off = 128; off > 0; off >>= 1) {
        if (t < off) red[t] += red[t + off];
        __syncthreads();
    }
    float inv = 1.0f / red[0];
#pragma unroll
    for (int k = 0; k < 16; ++k) prow[t + k * 256] = x[k] * inv;
}

// ---------------------------------------------------------------------------
// Stage 4: ctx partials.  ctx[n,h,e] = sum_s p[n,h,s]*src[s,n,e]  via WMMA.
// Grid (n, s-chunk of 512) -> 8 partials reduced afterwards (deterministic,
// no float atomics).  A = p rows (M=h pad16, K=s), B = src (K=s, N=e).
// ---------------------------------------------------------------------------
__global__ __launch_bounds__(256) void ctx_kernel(const float* __restrict__ src,
                                                  const float* __restrict__ p,
                                                  float* __restrict__ ctx_part) {
    int n = blockIdx.x;            // 0..31
    int c = blockIdx.y;            // 0..7   (512 s each)
    int wave = threadIdx.x >> 5;
    int lane = threadIdx.x & 31;
    int half = lane >> 4;
    int m = lane & 15;
    const float* prow = p + ((size_t)n * 16 + m) * SS + c * 512;  // padded heads

    v8f acc[4] = {{}, {}, {}, {}};
#pragma unroll
    for (int ks = 0; ks < 16; ++ks) {
        int sb0 = ks * 32;
        v16bf a = frag_contig(prow + sb0, half);
#pragma unroll
        for (int t4 = 0; t4 < 4; ++t4) {
            int e = (wave * 4 + t4) * 16 + m;   // this lane's column
            v16bf b = frag_src_strideS(src, c * 512 + sb0, n, e, half);
            acc[t4] = wmma_bf16(a, b, acc[t4]);
        }
    }
    if (half == 0) {
#pragma unroll
        for (int t4 = 0; t4 < 4; ++t4) {
            int e = (wave * 4 + t4) * 16 + m;
#pragma unroll
            for (int r = 0; r < 8; ++r)
                ctx_part[((size_t)c * (NN * HH) + (size_t)(n * HH + r)) * EE + e] = acc[t4][r];
        }
    }
}

__global__ __launch_bounds__(256) void ctxred_kernel(const float* __restrict__ part,
                                                     float* __restrict__ ctx) {
    int idx = blockIdx.x * 256 + threadIdx.x;   // 32*8*512 = 131072
    float acc = 0.0f;
#pragma unroll
    for (int c = 0; c < 8; ++c) acc += part[(size_t)c * 131072 + idx];
    ctx[idx] = acc;
}

// ---------------------------------------------------------------------------
// Stage 5: tiny output chain.
// mo[n,j] = Wv[j,:].ctx[n, j>>6, :] + bv[j]   (bv passes through: sum p == 1)
// o2 = mo @ Wo.T + bo ; out = o2 @ Wl.T + bl
// ---------------------------------------------------------------------------
__global__ __launch_bounds__(256) void vproj_kernel(const float* __restrict__ ctx,
                                                    const float* __restrict__ Wv,
                                                    const float* __restrict__ bv,
                                                    float* __restrict__ mo) {
    int idx = blockIdx.x * 256 + threadIdx.x;   // 16384
    int n = idx >> 9, j = idx & 511, h = j >> 6;
    const float* crow = ctx + ((size_t)n * HH + h) * EE;
    const float* wrow = Wv + (size_t)j * EE;
    float acc = bv[j];
#pragma unroll 4
    for (int e = 0; e < EE; ++e) acc = fmaf(crow[e], wrow[e], acc);
    mo[idx] = acc;
}

__global__ __launch_bounds__(256) void lin_kernel(const float* __restrict__ A,
                                                  const float* __restrict__ W,
                                                  const float* __restrict__ b,
                                                  float* __restrict__ out) {
    int idx = blockIdx.x * 256 + threadIdx.x;   // 16384
    int n = idx >> 9, j = idx & 511;
    const float* arow = A + (size_t)n * EE;
    const float* wrow = W + (size_t)j * EE;
    float acc = b[j];
#pragma unroll 4
    for (int e = 0; e < EE; ++e) acc = fmaf(arow[e], wrow[e], acc);
    out[idx] = acc;
}

// ---------------------------------------------------------------------------
extern "C" void kernel_launch(void* const* d_in, const int* in_sizes, int n_in,
                              void* d_out, int out_size, void* d_ws, size_t ws_size,
                              hipStream_t stream) {
    const float* src = (const float*)d_in[0];
    const float* Wq  = (const float*)d_in[1];
    const float* bq  = (const float*)d_in[2];
    const float* Wk  = (const float*)d_in[3];
    const float* bk  = (const float*)d_in[4];
    const float* Wv  = (const float*)d_in[5];
    const float* bv  = (const float*)d_in[6];
    const float* Wo  = (const float*)d_in[7];
    const float* bo  = (const float*)d_in[8];
    const float* Wl  = (const float*)d_in[9];
    const float* bl  = (const float*)d_in[10];

    float* ws = (float*)d_ws;
    float* q      = ws;               // 16384
    float* qkbuf  = ws + 16384;       // 32*16*512 = 262144 (head-padded)
    float* sb     = ws + 278528;      // 256
    float* scores = ws + 278784;      // 32*8*4096 = 1048576
    float* p      = ws + 1327360;     // 32*16*4096 = 2097152 (head-padded)
    float* part   = ws + 3424512;     // 8*32*8*512 = 1048576
    float* ctx    = ws + 4473088;     // 32*8*512 = 131072
    float* mo     = ws + 4604160;     // 16384
    float* o2     = ws + 4620544;     // 16384  (end: 4636928 floats ~ 17.7 MB)

    qproj_kernel  <<<64,   256, 0, stream>>>(src, Wq, bq, q);
    qk_kernel     <<<1024, 256, 0, stream>>>(q, Wk, bk, qkbuf, sb);
    scores_kernel <<<dim3(32, 16), 256, 0, stream>>>(src, qkbuf, scores);
    softmax_kernel<<<512,  256, 0, stream>>>(scores, sb, p);
    ctx_kernel    <<<dim3(32, 8),  256, 0, stream>>>(src, p, part);
    ctxred_kernel <<<512,  256, 0, stream>>>(part, ctx);
    vproj_kernel  <<<64,   256, 0, stream>>>(ctx, Wv, bv, mo);
    lin_kernel    <<<64,   256, 0, stream>>>(mo, Wo, bo, o2);
    lin_kernel    <<<64,   256, 0, stream>>>(o2, Wl, bl, (float*)d_out);
}
